// PerformerSelfAttention_71167608094795
// MI455X (gfx1250) — matrix operand order repaired and checked by reading, not compiled, not run on hardware
//
#include <hip/hip_runtime.h>
#include <hip/hip_bf16.h>

// ---------------------------------------------------------------------------
// Performer (FAVOR+) self-attention for gfx1250 / MI455X.
// B=4, L=4096, D=1024, H=16, DK=64, M_FEAT=256.
// bf16 WMMA (v_wmma_f32_16x16x32_bf16) for all GEMMs, f32 accumulation.
// Big GEMMs use the CDNA5 async-to-LDS pipeline (ASYNCcnt) with double
// buffering; fragments come from LDS via ds_load_b128.
// ---------------------------------------------------------------------------

#define BATCH   4
#define SEQ     4096
#define DMODEL  1024
#define NHEADS  16
#define DK      64
#define MFEAT   256
#define ROWS    (BATCH * SEQ)          // 16384

typedef __attribute__((ext_vector_type(16))) __bf16 v16bf;
typedef __attribute__((ext_vector_type(8)))  __bf16 v8bf;
typedef __attribute__((ext_vector_type(8)))  float  v8f;

// ---------------- bf16 helpers (bit-level, RNE) ----------------------------
__device__ __forceinline__ float bf2f(__bf16 x) {
    unsigned short s = __builtin_bit_cast(unsigned short, x);
    unsigned u = ((unsigned)s) << 16;
    return __builtin_bit_cast(float, u);
}
__device__ __forceinline__ __bf16 f2bf(float f) {
    unsigned u = __builtin_bit_cast(unsigned, f);
    unsigned r = u + 0x7FFFu + ((u >> 16) & 1u);
    unsigned short s = (unsigned short)(r >> 16);
    return __builtin_bit_cast(__bf16, s);
}

// ---------------- WMMA fragment helpers (per ISA 7.12.2 layouts) -----------
__device__ __forceinline__ v16bf cat8(v8bf a, v8bf b) {
    return __builtin_shufflevector(a, b, 0,1,2,3,4,5,6,7,8,9,10,11,12,13,14,15);
}

// A-matrix 16x32 bf16, row-major source, lda in elements.
// lane l: row = l&15; half=l>>4; elems 0..7 = K half*8.., elems 8..15 = K 16+half*8..
__device__ __forceinline__ v16bf load_a_frag(const __bf16* __restrict__ A,
                                             int lda, int row0, int k0, int lane) {
    int m = lane & 15, half = lane >> 4;
    const __bf16* p = A + (size_t)(row0 + m) * lda + k0 + half * 8;
    v8bf lo = *(const v8bf*)(p);
    v8bf hi = *(const v8bf*)(p + 16);
    return cat8(lo, hi);
}

// B-matrix 32x16 bf16. Source stored as [N rows][K cols] row-major (each output
// column is a contiguous row). lane l: col n = l&15; half=l>>4; elems 0..7 = K half*8.., elems 8..15 = K 16+half*8..
__device__ __forceinline__ v16bf load_b_frag(const __bf16* __restrict__ Bm,
                                             int ldb, int n0, int k0, int lane) {
    int n = lane & 15, half = lane >> 4;
    const __bf16* p = Bm + (size_t)(n0 + n) * ldb + k0 + half * 8;
    v8bf lo = *(const v8bf*)(p);
    v8bf hi = *(const v8bf*)(p + 16);
    return cat8(lo, hi);
}

__device__ __forceinline__ v8f wmma_bf16(v16bf a, v16bf b, v8f c) {
    return __builtin_amdgcn_wmma_f32_16x16x32_bf16(false, a, false, b,
                                                   (short)0, c, false, false);
}

// async copy of 16 bytes global -> LDS (ASYNCcnt-tracked, no VGPR staging)
__device__ __forceinline__ void async_g2l_b128(unsigned lds_off, const void* gptr) {
    asm volatile("global_load_async_to_lds_b128 %0, %1, off"
                 :: "v"(lds_off), "v"(gptr) : "memory");
}
__device__ __forceinline__ unsigned lds_addr(const void* p) {
    return (unsigned)(size_t)p;          // flat->LDS offset: low 32 bits
}

// ---------------- elementwise kernels --------------------------------------
__global__ void cvt_f32_bf16(const float* __restrict__ in, __bf16* __restrict__ out, int n) {
    int i = blockIdx.x * blockDim.x + threadIdx.x;
    if (i < n) out[i] = f2bf(in[i]);
}
__global__ void zero_f32(float* __restrict__ p, int n) {
    int i = blockIdx.x * blockDim.x + threadIdx.x;
    if (i < n) p[i] = 0.0f;
}
// kvF [H][MFEAT][80] f32 -> kvT [H][80][MFEAT] bf16
__global__ void kv_cvt(const float* __restrict__ kvF, __bf16* __restrict__ kvT) {
    int idx = blockIdx.x * blockDim.x + threadIdx.x;
    if (idx >= NHEADS * MFEAT * 80) return;
    int h = idx / (MFEAT * 80);
    int r = idx % (MFEAT * 80);
    int m = r / 80, c = r % 80;
    kvT[((size_t)h * 80 + c) * MFEAT + m] = f2bf(kvF[idx]);
}

// ---------------- big NT-GEMM: C[M,N] = A[M,K] * B[N,K]^T ------------------
// Async-to-LDS double-buffered pipeline. Block = 128 threads (4 waves),
// 64x64 C tile per block; wave w computes rows [w*16, w*16+16) x 64 cols.
// EPI 0: write bf16 head-split  [b][h][l][dk]   (Q, K)
// EPI 2: write bf16 transposed  [b][h][dk][l]   (V)
// EPI 1: write f32 row-major    [row][col]      (final output)
template<int EPI>
__global__ void gemm_nt(const __bf16* __restrict__ A, const __bf16* __restrict__ Bm,
                        void* __restrict__ Cout, int M, int N, int K) {
    // [buf][0]=A tile 64x32, [buf][1]=B tile 64x32 (bf16)  -> 16 KB total
    __shared__ __bf16 smem[2][2][64 * 32];

    int tid  = threadIdx.x;              // 0..127
    int lane = tid & 31;
    int wave = tid >> 5;                 // 0..3
    int row0 = blockIdx.x * 64;
    int col0 = blockIdx.y * 64;
    int wrow = row0 + wave * 16;

    // 512 b128 chunks per k-step (A:256, B:256); 4 per thread.
    // chunk c: isB=c>>8, rrow=(c&255)>>2 (tile row), part=(c&3) (16B unit).
    auto issue_copies = [&](int kk, int buf) {
        #pragma unroll
        for (int j = 0; j < 4; ++j) {
            int c    = tid + 128 * j;
            int isB  = c >> 8;
            int cc   = c & 255;
            int rrow = cc >> 2;
            int part = cc & 3;
            const __bf16* gp = (isB ? Bm + (size_t)(col0 + rrow) * K
                                    : A  + (size_t)(row0 + rrow) * K)
                               + kk + part * 8;
            async_g2l_b128(lds_addr(&smem[buf][isB][rrow * 32 + part * 8]), gp);
        }
    };

    v8f acc[4] = {};
    issue_copies(0, 0);
    int buf = 0;
    int m = lane & 15, half = lane >> 4;
    for (int k = 0; k < K; k += 32) {
        if (k + 32 < K) {
            issue_copies(k + 32, buf ^ 1);                 // prefetch next tile
            asm volatile("s_wait_asynccnt 0x4" ::: "memory"); // current tile done
        } else {
            asm volatile("s_wait_asynccnt 0x0" ::: "memory");
        }
        __syncthreads();                                   // visible to all waves

        const __bf16* As = &smem[buf][0][0];
        const __bf16* Bs = &smem[buf][1][0];
        v16bf afrag;
        {
            const __bf16* p = As + (wave * 16 + m) * 32 + half * 8;
            afrag = cat8(*(const v8bf*)p, *(const v8bf*)(p + 16));
        }
        #pragma unroll
        for (int t = 0; t < 4; ++t) {
            const __bf16* p = Bs + (t * 16 + m) * 32 + half * 8;
            v16bf bfrag = cat8(*(const v8bf*)p, *(const v8bf*)(p + 16));
            acc[t] = wmma_bf16(afrag, bfrag, acc[t]);
        }
        __syncthreads();                    // reads retired before next overwrite
        buf ^= 1;
    }

    #pragma unroll
    for (int t = 0; t < 4; ++t) {
        #pragma unroll
        for (int r = 0; r < 8; ++r) {
            int row = wrow + r + half * 8;
            int col = col0 + t * 16 + m;
            float v = acc[t][r];
            if (EPI == 1) {
                ((float*)Cout)[(size_t)row * N + col] = v;
            } else {
                int b  = row >> 12;              // L = 4096
                int l  = row & (SEQ - 1);
                int h  = col >> 6;               // DK = 64
                int dk = col & (DK - 1);
                __bf16 bv = f2bf(v);
                if (EPI == 0)
                    ((__bf16*)Cout)[(((size_t)b * NHEADS + h) * SEQ + l) * DK + dk] = bv;
                else
                    ((__bf16*)Cout)[(((size_t)b * NHEADS + h) * DK + dk) * SEQ + l] = bv;
            }
        }
    }
}

// ---------------- FAVOR+ feature map (VALU; ~5% of total FLOPs) ------------
// Input Qh: [H][L][DK] bf16 for one batch. proj: [MFEAT][DK] bf16 (staged in LDS).
// TRANSPOSED=true  -> phi_k, apply mask, write [H][MFEAT][L]
// TRANSPOSED=false -> phi_q,             write [H][L][MFEAT]
template<bool TRANSPOSED>
__global__ void favor_phi(const __bf16* __restrict__ Qh,
                          const __bf16* __restrict__ proj,
                          const int* __restrict__ mask,
                          __bf16* __restrict__ out) {
    __shared__ __bf16 sproj[MFEAT * DK];                   // 32 KB
    for (int i = threadIdx.x; i < MFEAT * DK / 8; i += blockDim.x)
        ((v8bf*)sproj)[i] = ((const v8bf*)proj)[i];
    __syncthreads();

    int tid = blockIdx.x * blockDim.x + threadIdx.x;       // [0, H*L)
    int h = tid >> 12;
    int l = tid & (SEQ - 1);
    const __bf16* q = Qh + ((size_t)h * SEQ + l) * DK;
    const float c = 0.35355339059327373f;                  // 64^-0.25
    float qf[DK];
    float nsq = 0.0f;
    #pragma unroll
    for (int k = 0; k < DK; ++k) { float v = c * bf2f(q[k]); qf[k] = v; nsq += v * v; }
    nsq *= 0.5f;
    // pass 1: row max of projection (recompute cheaper than 256-reg buffer)
    float pmax = -3.4e38f;
    for (int m = 0; m < MFEAT; ++m) {
        const __bf16* pr = sproj + m * DK;
        float a = 0.0f;
        #pragma unroll
        for (int k = 0; k < DK; ++k) a += qf[k] * bf2f(pr[k]);
        pmax = fmaxf(pmax, a);
    }
    float mval = 1.0f;
    if (TRANSPOSED) mval = (float)mask[l];
    const float ratio = 0.0625f;                           // 256^-0.5
    for (int m = 0; m < MFEAT; ++m) {
        const __bf16* pr = sproj + m * DK;
        float a = 0.0f;
        #pragma unroll
        for (int k = 0; k < DK; ++k) a += qf[k] * bf2f(pr[k]);
        float phi = ratio * (__expf(a - pmax - nsq) + 1e-6f) * mval;
        if (TRANSPOSED) out[((size_t)h * MFEAT + m) * SEQ + l] = f2bf(phi);
        else            out[((size_t)h * SEQ + l) * MFEAT + m] = f2bf(phi);
    }
}

// ---------------- kv = phi_k^T @ [V | 1] (split-K over L, atomic f32) ------
// phiKT: [H][MFEAT][L] bf16, VT: [H][DK][L] bf16, kvF: [H][MFEAT][80] f32
// grid (mTile=16, h=16, lsplit=8), 1 wave per block.
__global__ void kv_accum(const __bf16* __restrict__ phiKT,
                         const __bf16* __restrict__ VT,
                         float* __restrict__ kvF) {
    int lane = threadIdx.x & 31;
    int mt = blockIdx.x, h = blockIdx.y, ls = blockIdx.z;
    const __bf16* A  = phiKT + (size_t)h * MFEAT * SEQ;
    const __bf16* Bv = VT    + (size_t)h * DK * SEQ;
    v8f acc[5] = {};
    // "ones" B fragment: within-tile column 0 (global col 64) = 1.0 over all K
    v16bf onesB = {};
    {
        __bf16 val = ((lane & 15) == 0) ? f2bf(1.0f) : f2bf(0.0f);
        #pragma unroll
        for (int i = 0; i < 16; ++i) onesB[i] = val;
    }
    int k0 = ls * (SEQ / 8);
    for (int k = k0; k < k0 + SEQ / 8; k += 32) {
        v16bf a = load_a_frag(A, SEQ, mt * 16, k, lane);
        #pragma unroll
        for (int t = 0; t < 4; ++t) {
            v16bf b = load_b_frag(Bv, SEQ, t * 16, k, lane);
            acc[t] = wmma_bf16(a, b, acc[t]);
        }
        acc[4] = wmma_bf16(a, onesB, acc[4]);          // k_sum into col 64
    }
    int n = lane & 15, half = lane >> 4;
    #pragma unroll
    for (int t = 0; t < 5; ++t) {
        if (t == 4 && n != 0) continue;                // only col 64 meaningful
        #pragma unroll
        for (int r = 0; r < 8; ++r) {
            int m = mt * 16 + r + half * 8;
            int col = t * 16 + n;
            atomicAdd(&kvF[((size_t)h * MFEAT + m) * 80 + col], acc[t][r]);
        }
    }
}

// ---------------- out = (phi_q @ kv) / (phi_q . k_sum) ---------------------
// phiQ: [H][L][MFEAT] bf16, kvT: [H][80][MFEAT] bf16,
// attn: [L][DMODEL] bf16 (one batch). grid (lTile=256, h=16), 1 wave/block.
__global__ void attn_out(const __bf16* __restrict__ phiQ,
                         const __bf16* __restrict__ kvT,
                         __bf16* __restrict__ attn) {
    int lane = threadIdx.x & 31;
    int lt = blockIdx.x, h = blockIdx.y;
    const __bf16* A  = phiQ + (size_t)h * SEQ * MFEAT;
    const __bf16* Bm = kvT  + (size_t)h * 80 * MFEAT;
    v8f acc[5] = {};
    for (int k = 0; k < MFEAT; k += 32) {
        v16bf a = load_a_frag(A, MFEAT, lt * 16, k, lane);
        #pragma unroll
        for (int t = 0; t < 5; ++t) {
            v16bf b = load_b_frag(Bm, MFEAT, t * 16, k, lane);
            acc[t] = wmma_bf16(a, b, acc[t]);
        }
    }
    int n = lane & 15, half = lane >> 4;
    #pragma unroll
    for (int r = 0; r < 8; ++r) {
        // denominator = column 64 = tile 4, within-tile col 0 -> lane half*16
        float denom = __shfl(acc[4][r], lane & 16, 32);
        denom = fmaxf(denom, 1e-6f);
        int row = lt * 16 + r + half * 8;
        #pragma unroll
        for (int t = 0; t < 4; ++t) {
            int col = t * 16 + n;                       // dk 0..63
            attn[(size_t)row * DMODEL + h * DK + col] = f2bf(acc[t][r] / denom);
        }
    }
}

// ---------------------------------------------------------------------------
extern "C" void kernel_launch(void* const* d_in, const int* in_sizes, int n_in,
                              void* d_out, int out_size, void* d_ws, size_t ws_size,
                              hipStream_t stream) {
    const float* X    = (const float*)d_in[0];
    const int*   mask = (const int*)  d_in[1];
    const float* Wq   = (const float*)d_in[2];
    const float* Wk   = (const float*)d_in[3];
    const float* Wv   = (const float*)d_in[4];
    const float* Wo   = (const float*)d_in[5];
    const float* proj = (const float*)d_in[6];

    char* ws = (char*)d_ws;
    size_t off = 0;
    auto carve = [&](size_t bytes) {
        void* p = ws + off;
        off = (off + bytes + 255) & ~(size_t)255;
        return p;
    };
    __bf16* Xb    = (__bf16*)carve(sizeof(__bf16) * (size_t)ROWS * DMODEL);   // 32 MiB
    __bf16* Wqb   = (__bf16*)carve(sizeof(__bf16) * DMODEL * DMODEL);
    __bf16* Wkb   = (__bf16*)carve(sizeof(__bf16) * DMODEL * DMODEL);
    __bf16* Wvb   = (__bf16*)carve(sizeof(__bf16) * DMODEL * DMODEL);
    __bf16* Wob   = (__bf16*)carve(sizeof(__bf16) * DMODEL * DMODEL);
    __bf16* Projb = (__bf16*)carve(sizeof(__bf16) * MFEAT * DK);
    __bf16* Qb    = (__bf16*)carve(sizeof(__bf16) * (size_t)ROWS * DMODEL);   // [B][H][L][DK]
    __bf16* Kb    = (__bf16*)carve(sizeof(__bf16) * (size_t)ROWS * DMODEL);   // [B][H][L][DK]
    __bf16* VTb   = (__bf16*)carve(sizeof(__bf16) * (size_t)ROWS * DMODEL);   // [B][H][DK][L]
    __bf16* Phi   = (__bf16*)carve(sizeof(__bf16) * (size_t)NHEADS * MFEAT * SEQ); // reused
    float*  kvF   = (float*) carve(sizeof(float)  * NHEADS * MFEAT * 80);
    __bf16* kvT   = (__bf16*)carve(sizeof(__bf16) * NHEADS * 80 * MFEAT);
    __bf16* Attnb = Xb;   // alias: Xb dead after the V projection GEMM

    // 1) fp32 -> bf16 conversions
    {
        int n = ROWS * DMODEL;
        cvt_f32_bf16<<<(n + 255) / 256, 256, 0, stream>>>(X, Xb, n);
        n = DMODEL * DMODEL;
        cvt_f32_bf16<<<(n + 255) / 256, 256, 0, stream>>>(Wq, Wqb, n);
        cvt_f32_bf16<<<(n + 255) / 256, 256, 0, stream>>>(Wk, Wkb, n);
        cvt_f32_bf16<<<(n + 255) / 256, 256, 0, stream>>>(Wv, Wvb, n);
        cvt_f32_bf16<<<(n + 255) / 256, 256, 0, stream>>>(Wo, Wob, n);
        n = MFEAT * DK;
        cvt_f32_bf16<<<(n + 255) / 256, 256, 0, stream>>>(proj, Projb, n);
    }

    // 2) Q/K/V projections: 16384x1024x1024 bf16 WMMA GEMMs (async-LDS pipeline)
    dim3 ggrid(ROWS / 64, DMODEL / 64);
    gemm_nt<0><<<ggrid, 128, 0, stream>>>(Xb, Wqb, Qb,  ROWS, DMODEL, DMODEL);
    gemm_nt<0><<<ggrid, 128, 0, stream>>>(Xb, Wkb, Kb,  ROWS, DMODEL, DMODEL);
    gemm_nt<2><<<ggrid, 128, 0, stream>>>(Xb, Wvb, VTb, ROWS, DMODEL, DMODEL);

    // 3) per-batch linear attention (Phi buffer reused K-transposed then Q)
    for (int b = 0; b < BATCH; ++b) {
        const __bf16* Kb_b  = Kb  + (size_t)b * NHEADS * SEQ * DK;
        const __bf16* Qb_b  = Qb  + (size_t)b * NHEADS * SEQ * DK;
        const __bf16* VT_b  = VTb + (size_t)b * NHEADS * DK * SEQ;
        __bf16*       At_b  = Attnb + (size_t)b * SEQ * DMODEL;

        favor_phi<true><<<NHEADS * SEQ / 256, 256, 0, stream>>>(Kb_b, Projb,
                                                                mask + b * SEQ, Phi);
        {
            int n = NHEADS * MFEAT * 80;
            zero_f32<<<(n + 255) / 256, 256, 0, stream>>>(kvF, n);
        }
        kv_accum<<<dim3(MFEAT / 16, NHEADS, 8), 32, 0, stream>>>(Phi, VT_b, kvF);
        {
            int n = NHEADS * MFEAT * 80;
            kv_cvt<<<(n + 255) / 256, 256, 0, stream>>>(kvF, kvT);
        }
        favor_phi<false><<<NHEADS * SEQ / 256, 256, 0, stream>>>(Qb_b, Projb,
                                                                 nullptr, Phi);
        attn_out<<<dim3(SEQ / 16, NHEADS), 32, 0, stream>>>(Phi, kvT, At_b);
    }

    // 4) output projection -> d_out (fp32)
    gemm_nt<1><<<ggrid, 128, 0, stream>>>(Attnb, Wob, d_out, ROWS, DMODEL, DMODEL);
}
